// PriorConditionedSelectiveStateSpace_53446573031746
// MI455X (gfx1250) — matrix operand order repaired and checked by reading, not compiled, not run on hardware
//
#include <hip/hip_runtime.h>
#include <math.h>

// ---------------- types / constants ----------------
typedef __attribute__((ext_vector_type(16))) __bf16 v16bf;
typedef __attribute__((ext_vector_type(8)))  float  v8f;

union BFrag { v16bf v; unsigned short h[16]; unsigned u[8]; };
union CFrag { v8f   v; float f[8]; };

constexpr int kB   = 8;     // batch
constexpr int kCIN = 64;
constexpr int kHID = 128;
constexpr int kH   = 128;
constexpr int kW   = 128;

constexpr int XSTR  = 72;   // padded row stride (ushorts) for x_bf   (bank-conflict-free)
constexpr int PSTR  = 136;  // padded row stride (ushorts) for 128-ch LDS tiles

// fragment table: W_in -> frags [0,16), W_delta -> [16,48), W_B -> [48,80), W_out -> [80,96)
constexpr int NFRAG = 96;

// ---------------- bf16 helpers (native cvt) ----------------
__device__ __forceinline__ unsigned short f2bf(float f) {
    union { __bf16 h; unsigned short s; } c;
    c.h = (__bf16)f;                       // native v_cvt to bf16
    return c.s;
}
__device__ __forceinline__ float bf2f(unsigned short s) {
    union { unsigned short s; __bf16 h; } c;
    c.s = s;
    return (float)c.h;
}
__device__ __forceinline__ unsigned pack2(float a, float b) {
    return (unsigned)f2bf(a) | ((unsigned)f2bf(b) << 16);
}

// ---------------- kernel 0: swizzle weights into WMMA A-fragments (bf16) ----------------
// A-fragment k order used here (16x32 bf16): lane<16 holds row M=lane, elements e:
//   e=0..15 -> K=e; lane>=16: e=0..15 -> K=16+e (contiguous, matches B loads).
__global__ void k_prep(const float* Wi, const float* Wd, const float* Wb, const float* Wo,
                       unsigned short* fragbuf)
{
    int frag = blockIdx.x;      // 0..95
    int lane = threadIdx.x;     // 0..31
    const float* Wm; int cols, mt, kt;
    if (frag < 16)      { int f = frag;      Wm = Wi; cols = 64;  mt = f >> 1; kt = f & 1; }
    else if (frag < 48) { int f = frag - 16; Wm = Wd; cols = 128; mt = f >> 2; kt = f & 3; }
    else if (frag < 80) { int f = frag - 48; Wm = Wb; cols = 128; mt = f >> 2; kt = f & 3; }
    else                { int f = frag - 80; Wm = Wo; cols = 128; mt = f >> 2; kt = f & 3; }
    int m    = mt * 16 + (lane & 15);
    int base = (lane < 16) ? 0 : 8;
    unsigned short* dst = fragbuf + ((size_t)frag * 32 + lane) * 16;
#pragma unroll
    for (int e = 0; e < 16; ++e) {
        int k = kt * 32 + 2 * base + e;
        dst[e] = f2bf(Wm[(size_t)m * cols + k]);
    }
}

// ---------------- kernel 1: bilinear(align_corners) upsample + clip ----------------
__global__ void k_prior(const float* __restrict__ prior, float* __restrict__ pu_ws,
                        float* __restrict__ pu_out)
{
    int idx = blockIdx.x * 256 + threadIdx.x;
    if (idx >= kB * kH * kW) return;
    int w = idx & 127, h = (idx >> 7) & 127, b = idx >> 14;
    const float sc = 63.0f / 127.0f;
    float fy = h * sc, fx = w * sc;
    int y0 = (int)floorf(fy), x0 = (int)floorf(fx);
    int y1 = y0 + 1 < 63 ? y0 + 1 : 63;
    int x1 = x0 + 1 < 63 ? x0 + 1 : 63;
    float wy = fy - (float)y0, wx = fx - (float)x0;
    const float* p = prior + (size_t)b * 64 * 64;
    float r0 = p[y0 * 64 + x0] * (1.0f - wy) + p[y1 * 64 + x0] * wy;
    float r1 = p[y0 * 64 + x1] * (1.0f - wy) + p[y1 * 64 + x1] * wy;
    float v  = r0 * (1.0f - wx) + r1 * wx;
    v = fminf(fmaxf(v, -1.0f), 1.0f);
    pu_ws[idx]  = v;
    pu_out[idx] = v;
}

// ---------------- kernel 2: fused front GEMMs + gating; emit A_bar, b (bf16, [b][h][w][c]) ----
__global__ __launch_bounds__(256) void k_front(
    const float* __restrict__ x, const float* __restrict__ pu_ws,
    const unsigned short* __restrict__ fragbuf,
    const float* __restrict__ b_in, const float* __restrict__ b_delta,
    const float* __restrict__ b_B,
    const float* __restrict__ lambda_p, const float* __restrict__ alpha_p,
    const float* __restrict__ A,
    unsigned short* __restrict__ Abar, unsigned short* __restrict__ bvec)
{
    __shared__ unsigned short x_bf[kW * XSTR];    // [w][c] bf16, padded
    __shared__ unsigned short xp_bf[kW * PSTR];   // [w][m] bf16, padded
    __shared__ float pu_s[kW];
    __shared__ float negA_s[kHID], bin_s[kHID], bd_s[kHID], bb_s[kHID];

    const int tid = threadIdx.x;
    const int b = blockIdx.x >> 7, h = blockIdx.x & 127;

    // stage x row (coalesced along w)
    const float* xrow = x + (size_t)b * kCIN * kH * kW + (size_t)h * kW;
    for (int i = tid; i < kCIN * kW; i += 256) {
        int c = i >> 7, w = i & 127;
        x_bf[w * XSTR + c] = f2bf(xrow[(size_t)c * kH * kW + w]);
    }
    if (tid < kW)  pu_s[tid] = pu_ws[(size_t)b * kH * kW + (size_t)h * kW + tid];
    if (tid < kHID) {
        negA_s[tid] = -expf(A[tid]);
        bin_s[tid] = b_in[tid]; bd_s[tid] = b_delta[tid]; bb_s[tid] = b_B[tid];
    }
    __syncthreads();

    const int lane = tid & 31, wave = tid >> 5;
    const int n = wave * 16 + (lane & 15);   // pixel (w) owned by this lane/tile
    const int khalf = lane >> 4;
    const int rowoff = khalf * 8;            // C-fragment M offset for this lane half
    const float lambda = lambda_p[0], alpha = alpha_p[0];

    // ---- GEMM1: x_proj = W_in @ x + b_in  (M=128, K=64, N=16 per wave)
    CFrag accP[8];
#pragma unroll
    for (int mt = 0; mt < 8; ++mt) {
        CFrag c;
#pragma unroll
        for (int r = 0; r < 8; ++r) c.f[r] = bin_s[mt * 16 + rowoff + r];
#pragma unroll
        for (int kt = 0; kt < 2; ++kt) {
            BFrag a; a.v = *(const v16bf*)(fragbuf + ((size_t)(mt * 2 + kt) * 32 + lane) * 16);
            BFrag bb; bb.v = *(const v16bf*)(&x_bf[n * XSTR + kt * 32 + khalf * 16]);
            c.v = __builtin_amdgcn_wmma_f32_16x16x32_bf16(false, a.v, false, bb.v,
                                                          (short)0, c.v, false, false);
        }
        accP[mt] = c;
#pragma unroll
        for (int r = 0; r < 8; r += 2)
            *(unsigned*)&xp_bf[n * PSTR + mt * 16 + rowoff + r] = pack2(c.f[r], c.f[r + 1]);
    }
    __syncthreads();

    // ---- GEMM2/3: delta_pre = W_delta @ x_proj + b_delta ; B_pre = W_B @ x_proj + b_B
    const float pu = pu_s[n];
    const size_t gbase = (size_t)blockIdx.x * (kW * kHID);
#pragma unroll
    for (int mt = 0; mt < 8; ++mt) {
        CFrag cd, cb;
#pragma unroll
        for (int r = 0; r < 8; ++r) {
            cd.f[r] = bd_s[mt * 16 + rowoff + r];
            cb.f[r] = bb_s[mt * 16 + rowoff + r];
        }
#pragma unroll
        for (int kt = 0; kt < 4; ++kt) {
            BFrag bb; bb.v = *(const v16bf*)(&xp_bf[n * PSTR + kt * 32 + khalf * 16]);
            BFrag ad; ad.v = *(const v16bf*)(fragbuf + ((size_t)(16 + mt * 4 + kt) * 32 + lane) * 16);
            BFrag ab; ab.v = *(const v16bf*)(fragbuf + ((size_t)(48 + mt * 4 + kt) * 32 + lane) * 16);
            cd.v = __builtin_amdgcn_wmma_f32_16x16x32_bf16(false, ad.v, false, bb.v,
                                                           (short)0, cd.v, false, false);
            cb.v = __builtin_amdgcn_wmma_f32_16x16x32_bf16(false, ab.v, false, bb.v,
                                                           (short)0, cb.v, false, false);
        }
        float at[8], bt[8];
#pragma unroll
        for (int r = 0; r < 8; ++r) {
            int m = mt * 16 + rowoff + r;
            float dpre  = cd.f[r] + lambda * pu;
            float delta = (dpre > 20.0f) ? dpre : log1pf(expf(dpre));       // softplus
            at[r] = expf(delta * negA_s[m]);                                // A_bar
            float Bk = cb.f[r] * (1.0f + alpha * pu);
            bt[r] = delta * Bk * accP[mt].f[r];                             // B_bar * feat
        }
#pragma unroll
        for (int r = 0; r < 8; r += 2) {
            size_t o = gbase + (size_t)n * kHID + mt * 16 + rowoff + r;
            *(unsigned*)&Abar[o] = pack2(at[r], at[r + 1]);
            *(unsigned*)&bvec[o] = pack2(bt[r], bt[r + 1]);
        }
    }
}

// ---------------- kernel 3: horizontal scan + W_out GEMM, write base output ----------
__global__ __launch_bounds__(256) void k_row(
    const unsigned short* __restrict__ Abar, const unsigned short* __restrict__ bvec,
    const unsigned short* __restrict__ fragbuf,
    const float* __restrict__ x, const float* __restrict__ b_out,
    const float* __restrict__ gamma_p, float* __restrict__ out)
{
    __shared__ unsigned short s_s[kW * PSTR];   // scan_h as bf16 [w][c]
    const int tid = threadIdx.x;
    const int b = blockIdx.x >> 7, h = blockIdx.x & 127;
    const size_t gbase = (size_t)blockIdx.x * (kW * kHID);

    if (tid < kHID) {                           // one channel per thread, serial over w
        int c = tid;
        float p = 1.0f, s = 0.0f;
        for (int w = 0; w < kW; ++w) {
            size_t o = gbase + (size_t)w * kHID + c;   // coalesced across threads
            if (w + 8 < kW) {                          // stream prefetch 8 pixels ahead
                __builtin_prefetch(&Abar[o + 8 * kHID], 0, 0);
                __builtin_prefetch(&bvec[o + 8 * kHID], 0, 0);
            }
            p *= bf2f(Abar[o]);
            float ps = fmaxf(p, 1e-8f);
            s += bf2f(bvec[o]) / ps;
            s_s[w * PSTR + c] = f2bf(p * s);
        }
    }
    __syncthreads();

    const int lane = tid & 31, wave = tid >> 5;
    const int n = wave * 16 + (lane & 15);
    const int khalf = lane >> 4, rowoff = khalf * 8;
    const float gamma = gamma_p[0];
#pragma unroll
    for (int mt = 0; mt < 4; ++mt) {            // M = 64 output channels
        CFrag c;
#pragma unroll
        for (int r = 0; r < 8; ++r) c.f[r] = 0.0f;
#pragma unroll
        for (int kt = 0; kt < 4; ++kt) {
            BFrag a; a.v = *(const v16bf*)(fragbuf + ((size_t)(80 + mt * 4 + kt) * 32 + lane) * 16);
            BFrag bb; bb.v = *(const v16bf*)(&s_s[n * PSTR + kt * 32 + khalf * 16]);
            c.v = __builtin_amdgcn_wmma_f32_16x16x32_bf16(false, a.v, false, bb.v,
                                                          (short)0, c.v, false, false);
        }
#pragma unroll
        for (int r = 0; r < 8; ++r) {
            int o = mt * 16 + rowoff + r;
            size_t oi = (((size_t)b * kCIN + o) * kH + h) * kW + n;
            out[oi] = x[oi] + gamma * (b_out[o] + c.f[r]);
        }
    }
}

// ---------------- kernel 4: vertical scan + W_out GEMM, atomic accumulate -------------
__global__ __launch_bounds__(256) void k_col(
    const unsigned short* __restrict__ Abar, const unsigned short* __restrict__ bvec,
    const unsigned short* __restrict__ fragbuf,
    const float* __restrict__ gamma_p, float* __restrict__ out)
{
    __shared__ unsigned short s_s[kH * PSTR];   // scan_v as bf16 [h][c]
    const int tid = threadIdx.x;
    const int b = blockIdx.x >> 7, w = blockIdx.x & 127;

    if (tid < kHID) {
        int c = tid;
        float p = 1.0f, s = 0.0f;
        for (int h = 0; h < kH; ++h) {
            size_t o = (((size_t)b * kH + h) * kW + w) * kHID + c;  // 256B coalesced chunk
            if (h + 8 < kH) {                                       // prefetch 8 rows ahead
                __builtin_prefetch(&Abar[o + (size_t)8 * kW * kHID], 0, 0);
                __builtin_prefetch(&bvec[o + (size_t)8 * kW * kHID], 0, 0);
            }
            p *= bf2f(Abar[o]);
            float ps = fmaxf(p, 1e-8f);
            s += bf2f(bvec[o]) / ps;
            s_s[h * PSTR + c] = f2bf(p * s);
        }
    }
    __syncthreads();

    const int lane = tid & 31, wave = tid >> 5;
    const int n = wave * 16 + (lane & 15);      // n indexes h here
    const int khalf = lane >> 4, rowoff = khalf * 8;
    const float gamma = gamma_p[0];
#pragma unroll
    for (int mt = 0; mt < 4; ++mt) {
        CFrag c;
#pragma unroll
        for (int r = 0; r < 8; ++r) c.f[r] = 0.0f;
#pragma unroll
        for (int kt = 0; kt < 4; ++kt) {
            BFrag a; a.v = *(const v16bf*)(fragbuf + ((size_t)(80 + mt * 4 + kt) * 32 + lane) * 16);
            BFrag bb; bb.v = *(const v16bf*)(&s_s[n * PSTR + kt * 32 + khalf * 16]);
            c.v = __builtin_amdgcn_wmma_f32_16x16x32_bf16(false, a.v, false, bb.v,
                                                          (short)0, c.v, false, false);
        }
#pragma unroll
        for (int r = 0; r < 8; ++r) {
            int o = mt * 16 + rowoff + r;
            size_t oi = (((size_t)b * kCIN + o) * kH + n) * kW + w;
            atomicAdd(&out[oi], gamma * c.f[r]);
        }
    }
}

// ---------------- launcher ----------------
extern "C" void kernel_launch(void* const* d_in, const int* in_sizes, int n_in,
                              void* d_out, int out_size, void* d_ws, size_t ws_size,
                              hipStream_t stream)
{
    const float* x        = (const float*)d_in[0];
    const float* prior    = (const float*)d_in[1];
    const float* W_in     = (const float*)d_in[2];
    const float* b_in     = (const float*)d_in[3];
    const float* W_out    = (const float*)d_in[4];
    const float* b_out    = (const float*)d_in[5];
    const float* W_delta  = (const float*)d_in[6];
    const float* b_delta  = (const float*)d_in[7];
    const float* W_B      = (const float*)d_in[8];
    const float* b_B      = (const float*)d_in[9];
    const float* lambda_d = (const float*)d_in[10];
    const float* alpha    = (const float*)d_in[11];
    const float* A        = (const float*)d_in[12];
    const float* gamma    = (const float*)d_in[13];
    float* out = (float*)d_out;

    // workspace layout
    char* ws = (char*)d_ws;
    size_t off = 0;
    float* pu_ws = (float*)(ws + off);               off += (size_t)kB * kH * kW * 4;        // 512KB
    unsigned short* fragbuf = (unsigned short*)(ws + off); off += (size_t)NFRAG * 32 * 16 * 2; // 96KB
    off = (off + 255) & ~(size_t)255;
    unsigned short* AbarB = (unsigned short*)(ws + off); off += (size_t)kB * kH * kW * kHID * 2; // 32MB
    unsigned short* bvecB = (unsigned short*)(ws + off); off += (size_t)kB * kH * kW * kHID * 2; // 32MB
    (void)ws_size; (void)n_in; (void)in_sizes; (void)out_size;

    float* prior_out = out + (size_t)kB * kCIN * kH * kW;   // tuple second element

    k_prep <<<NFRAG, 32, 0, stream>>>(W_in, W_delta, W_B, W_out, fragbuf);
    k_prior<<<(kB * kH * kW + 255) / 256, 256, 0, stream>>>(prior, pu_ws, prior_out);
    k_front<<<kB * kH, 256, 0, stream>>>(x, pu_ws, fragbuf, b_in, b_delta, b_B,
                                         lambda_d, alpha, A, AbarB, bvecB);
    k_row  <<<kB * kH, 256, 0, stream>>>(AbarB, bvecB, fragbuf, x, b_out, gamma, out);
    k_col  <<<kB * kW, 256, 0, stream>>>(AbarB, bvecB, fragbuf, gamma, out);
}
